// InterframeEncoder_51573967290911
// MI455X (gfx1250) — matrix operand order repaired and checked by reading, not compiled, not run on hardware
//
#include <hip/hip_runtime.h>

typedef __attribute__((ext_vector_type(16))) _Float16 v16h;
typedef __attribute__((ext_vector_type(8)))  _Float16 v8h;
typedef __attribute__((ext_vector_type(8)))  float    v8f;
typedef __attribute__((ext_vector_type(4)))  float    v4f;

#define ZH8 v8h{(_Float16)0,(_Float16)0,(_Float16)0,(_Float16)0,(_Float16)0,(_Float16)0,(_Float16)0,(_Float16)0}

// ---------------------------------------------------------------------------
// Weight packing: W is a flat [Ktot, Cout] f32 matrix (DHWIO). Pack into WMMA
// B-fragment order: halves [((chunk*n16s+n16)*32 + lane)*16 + j], lane 0-15
// holds K={0..7,16..23} (col N=lane), lane 16-31 holds K={8..15,24..31} (N=lane-16).
// ---------------------------------------------------------------------------
__global__ __launch_bounds__(256) void pack_w_kernel(
    const float* __restrict__ w, _Float16* __restrict__ pk,
    int Ktot, int Cout, int total /*chunks*n16s*/, int n16s)
{
    int idx = blockIdx.x * blockDim.x + threadIdx.x;
    if (idx >= total * 512) return;
    int j    = idx & 15;
    int lane = (idx >> 4) & 31;
    int t    = idx >> 9;              // t = chunk*n16s + n16
    int n16   = t % n16s;
    int chunk = t / n16s;
    int hi = lane >> 4;
    int kg = chunk * 32 + (hi << 3) + j + ((j >= 8) ? 8 : 0);
    int co = (n16 << 4) + (lane & 15);
    float v = (kg < Ktot && co < Cout) ? w[(size_t)kg * Cout + co] : 0.0f;
    pk[idx] = (_Float16)v;
}

__global__ __launch_bounds__(256) void pool_mask_kernel(
    const float* __restrict__ mi, float* __restrict__ mo, int Dout)
{
    int idx = blockIdx.x * blockDim.x + threadIdx.x;
    int n = Dout * Dout * Dout;
    if (idx >= n) return;
    int x = idx % Dout;
    int y = (idx / Dout) % Dout;
    int z = idx / (Dout * Dout);
    int Din = Dout * 2;
    float m = 0.0f;
    for (int dz = 0; dz < 2; ++dz)
        for (int dy = 0; dy < 2; ++dy)
            for (int dx = 0; dx < 2; ++dx) {
                size_t o = (((size_t)(2*z+dz) * Din) + (2*y+dy)) * Din + (2*x+dx);
                m = fmaxf(m, mi[o]);
            }
    mo[idx] = m;
}

__global__ __launch_bounds__(256) void f32_to_f16_kernel(
    const float* __restrict__ in, _Float16* __restrict__ out, size_t n)
{
    size_t i = (size_t)blockIdx.x * blockDim.x + threadIdx.x;
    if (i < n) out[i] = (_Float16)in[i];
}

__global__ __launch_bounds__(256) void f16_to_f32_kernel(
    const _Float16* __restrict__ in, float* __restrict__ out, size_t n)
{
    size_t i = (size_t)blockIdx.x * blockDim.x + threadIdx.x;
    if (i < n) out[i] = (float)in[i];
}

// Shared epilogue: (acc + bias) * mask, optional relu, optional residual add.
// C/D fragment layout: lane column N = lane&15, rows M = r + 8*(lane>=16).
template <int DOUT, int COUT, int CSTRIDE, int CHOFS, int RELU, bool HASRES>
__device__ __forceinline__ void conv_epilogue(
    const v8f& acc, const float* __restrict__ bias,
    const float* __restrict__ mask, _Float16* __restrict__ out,
    const _Float16* __restrict__ res,
    int z0, int y0, int tx, int lane, int n16)
{
    int hi = lane >> 4;
    int col = (n16 << 4) + (lane & 15);
    float bval = (col < COUT) ? bias[col] : 0.0f;
    int rowBase = ((z0 * DOUT + y0) * DOUT) + (tx << 4);
    const float* mrow = mask + rowBase + (hi << 3);
    v4f mlo = *(const v4f*)mrow;
    v4f mhi = *(const v4f*)(mrow + 4);
    size_t obase = ((size_t)rowBase + (hi << 3)) * CSTRIDE + CHOFS + col;
#pragma unroll
    for (int r = 0; r < 8; ++r) {
        float mval = (r < 4) ? mlo[r & 3] : mhi[r & 3];
        float v = (acc[r] + bval) * mval;
        if (RELU) v = fmaxf(v, 0.0f);
        if (col < COUT) {
            size_t oidx = obase + (size_t)r * CSTRIDE;   // imm offsets
            if (HASRES) v += (float)res[oidx];
            out[oidx] = (_Float16)v;
        }
    }
}

// ---------------------------------------------------------------------------
// Implicit-GEMM conv via V_WMMA_F32_16X16X32_F16, fully compile-time shape:
//   KS=3 -> stride1/pad1 SAME; KS=2 -> stride2/pad0 VALID; KS=1.
//   Taps-per-chunk derived from CIN: >=32 -> 1 (CIN/32 chunks/tap),
//   16 -> 2 taps/chunk, 8 -> 4 taps/chunk.
// All A/B offsets are literals -> immediate-offset b128 loads; all index
// math is shifts (power-of-two dims). A-fragment per ISA layout: lane<16
// row M holds K {0..7,16..23}; lane>=16 holds K {8..15,24..31}.
// ---------------------------------------------------------------------------
template <int KS, int DIN, int CIN, int COUT, int CSTRIDE, int CHOFS,
          int RELU, bool HASRES>
__global__ __launch_bounds__(256) void conv3d_wmma_kernel(
    const _Float16* __restrict__ in, const _Float16* __restrict__ wpk,
    const float* __restrict__ bias, const float* __restrict__ mask,
    _Float16* __restrict__ out, const _Float16* __restrict__ res)
{
    constexpr int STRD = (KS == 2) ? 2 : 1;
    constexpr int PAD  = (KS == 3) ? 1 : 0;
    constexpr int NT   = KS * KS * KS;
    constexpr int DOUT = (KS == 2) ? DIN / 2 : DIN;
    constexpr int N16S = (COUT + 15) / 16;
    constexpr int TILESX = DOUT / 16;
    constexpr long TOTALW = (long)DOUT * DOUT * TILESX * N16S;
    constexpr int TPC = (CIN >= 32) ? 1 : ((CIN == 16) ? 2 : 4);
    constexpr ptrdiff_t CINS = CIN;
    constexpr ptrdiff_t ROWS = (ptrdiff_t)DIN * CIN;
    constexpr ptrdiff_t SLIS = (ptrdiff_t)DIN * DIN * CIN;
    constexpr size_t WSTEP = (size_t)N16S * 512;

    long wid = ((long)blockIdx.x * blockDim.x + threadIdx.x) >> 5;
    int lane = threadIdx.x & 31;
    if (wid >= TOTALW) return;

    int n16 = (int)(wid % N16S);        // power-of-two -> and/shift
    long sp = wid / N16S;
    int tx = (int)(sp % TILESX);
    int y0 = (int)((sp / TILESX) % DOUT);
    int z0 = (int)(sp / ((long)TILESX * DOUT));
    int M  = lane & 15;
    int hi = lane >> 4;
    int vx = (tx << 4) + M;

    const int bz = z0 * STRD - PAD;
    const int by = y0 * STRD - PAD;
    const int bx = vx * STRD - PAD;
    const _Float16* vbase = in + (ptrdiff_t)bz * SLIS + (ptrdiff_t)by * ROWS
                               + (ptrdiff_t)bx * CINS;
    const _Float16* wbase = wpk + ((size_t)n16 * 32 + lane) * 16;
    __builtin_prefetch((const void*)wbase, 0, 0);

    v8f acc = {0.f, 0.f, 0.f, 0.f, 0.f, 0.f, 0.f, 0.f};

    if constexpr (TPC == 1) {
        // CIN >= 32: each tap contributes CIN/32 chunks; ci offset 8*hi.
        constexpr int CC = CIN / 32;
#pragma unroll
        for (int t = 0; t < NT; ++t) {
            constexpr int KS2 = KS * KS;
            const int dz = t / KS2, dy = (t / KS) % KS, dx = t % KS;
            bool ok = true;
            if constexpr (KS == 3) {
                ok = (unsigned)(bz + dz) < (unsigned)DIN &&
                     (unsigned)(by + dy) < (unsigned)DIN &&
                     (unsigned)(bx + dx) < (unsigned)DIN;
            }
            const _Float16* base = vbase + dz * SLIS + dy * ROWS + dx * CINS
                                         + (hi << 3);
#pragma unroll
            for (int cc = 0; cc < CC; ++cc) {
                const int chunk = t * CC + cc;
                v16h B = *(const v16h*)(wbase + chunk * WSTEP);
                v8h a0 = ZH8, a1 = ZH8;
                if (ok) {
                    a0 = *(const v8h*)(base + cc * 32);
                    a1 = *(const v8h*)(base + cc * 32 + 16);
                }
                v16h A = __builtin_shufflevector(a0, a1,
                    0,1,2,3,4,5,6,7,8,9,10,11,12,13,14,15);
                acc = __builtin_amdgcn_wmma_f32_16x16x32_f16(
                    false, A, false, B, (short)0, acc, false, false);
            }
        }
    } else if constexpr (TPC == 2) {
        // CIN == 16: chunk c covers taps {2c, 2c+1}; ci offset 8*hi each.
        constexpr int CH = (NT + 1) / 2;
#pragma unroll
        for (int c = 0; c < CH; ++c) {
            v16h B = *(const v16h*)(wbase + c * WSTEP);
            v8h a0 = ZH8, a1 = ZH8;
            {
                const int t = 2 * c;
                const int dz = t / (KS*KS), dy = (t / KS) % KS, dx = t % KS;
                bool ok = true;
                if constexpr (KS == 3) {
                    ok = (unsigned)(bz + dz) < (unsigned)DIN &&
                         (unsigned)(by + dy) < (unsigned)DIN &&
                         (unsigned)(bx + dx) < (unsigned)DIN;
                }
                if (ok) a0 = *(const v8h*)(vbase + dz * SLIS + dy * ROWS
                                                 + dx * CINS + (hi << 3));
            }
            {
                const int t = 2 * c + 1;
                if (t < NT) {   // literal -> pruned for the padded tail
                    const int dz = t / (KS*KS), dy = (t / KS) % KS, dx = t % KS;
                    bool ok = true;
                    if constexpr (KS == 3) {
                        ok = (unsigned)(bz + dz) < (unsigned)DIN &&
                             (unsigned)(by + dy) < (unsigned)DIN &&
                             (unsigned)(bx + dx) < (unsigned)DIN;
                    }
                    if (ok) a1 = *(const v8h*)(vbase + dz * SLIS + dy * ROWS
                                                     + dx * CINS + (hi << 3));
                }
            }
            v16h A = __builtin_shufflevector(a0, a1,
                0,1,2,3,4,5,6,7,8,9,10,11,12,13,14,15);
            acc = __builtin_amdgcn_wmma_f32_16x16x32_f16(
                false, A, false, B, (short)0, acc, false, false);
        }
    } else {
        // CIN == 8: chunk c covers taps {4c..4c+3}; group g: lo lanes take
        // tap 4c+2g, hi lanes tap 4c+2g+1 (each load = full Cin=8).
        constexpr int CH = (NT + 3) / 4;
#pragma unroll
        for (int c = 0; c < CH; ++c) {
            v16h B = *(const v16h*)(wbase + c * WSTEP);
            v8h ag[2];
#pragma unroll
            for (int g = 0; g < 2; ++g) {
                const int ta = 4 * c + 2 * g;
                const int tb = ta + 1;
                ptrdiff_t offA = 0; bool okA = false;
                if (ta < NT) {
                    const int dz = ta / (KS*KS), dy = (ta / KS) % KS, dx = ta % KS;
                    offA = dz * SLIS + dy * ROWS + dx * CINS;
                    okA = true;
                    if constexpr (KS == 3) {
                        okA = (unsigned)(bz + dz) < (unsigned)DIN &&
                              (unsigned)(by + dy) < (unsigned)DIN &&
                              (unsigned)(bx + dx) < (unsigned)DIN;
                    }
                }
                ptrdiff_t offB = 0; bool okB = false;
                if (tb < NT) {
                    const int dz = tb / (KS*KS), dy = (tb / KS) % KS, dx = tb % KS;
                    offB = dz * SLIS + dy * ROWS + dx * CINS;
                    okB = true;
                    if constexpr (KS == 3) {
                        okB = (unsigned)(bz + dz) < (unsigned)DIN &&
                              (unsigned)(by + dy) < (unsigned)DIN &&
                              (unsigned)(bx + dx) < (unsigned)DIN;
                    }
                }
                ptrdiff_t off = hi ? offB : offA;   // select of two literals
                bool ok = hi ? okB : okA;
                v8h a = ZH8;
                if (ok) a = *(const v8h*)(vbase + off);
                ag[g] = a;
            }
            v16h A = __builtin_shufflevector(ag[0], ag[1],
                0,1,2,3,4,5,6,7,8,9,10,11,12,13,14,15);
            acc = __builtin_amdgcn_wmma_f32_16x16x32_f16(
                false, A, false, B, (short)0, acc, false, false);
        }
    }

    conv_epilogue<DOUT, COUT, CSTRIDE, CHOFS, RELU, HASRES>(
        acc, bias, mask, out, res, z0, y0, tx, lane, n16);
}

// ---------------------------------------------------------------------------
// conv0: 3x3x3 SAME, Cin=1, Cout=16, D=128. Single K-chunk (27 taps + pad).
// ---------------------------------------------------------------------------
template <int D>
__global__ __launch_bounds__(256) void conv0_wmma_kernel(
    const _Float16* __restrict__ in, const _Float16* __restrict__ wpk,
    const float* __restrict__ bias, const float* __restrict__ mask,
    _Float16* __restrict__ out)
{
    constexpr int TILESX = D / 16;
    constexpr long TOTALW = (long)D * D * TILESX;
    long wid = ((long)blockIdx.x * blockDim.x + threadIdx.x) >> 5;
    int lane = threadIdx.x & 31;
    if (wid >= TOTALW) return;

    int tx = (int)(wid % TILESX);
    int y0 = (int)((wid / TILESX) % D);
    int z0 = (int)(wid / ((long)TILESX * D));
    int M  = lane & 15;
    int hi = lane >> 4;
    int vx = (tx << 4) + M;

    const _Float16* vbase = in + ((ptrdiff_t)(z0 - 1) * D + (y0 - 1)) * D + (vx - 1);

    v16h A;
#pragma unroll
    for (int j = 0; j < 16; ++j) {
        const int t0 = (j < 8) ? j : j + 8;   // literal: lo-lane tap
        const int t1 = t0 + 8;                // literal: hi-lane tap
        int dz = hi ? (t1 / 9) : (t0 / 9);
        int dy = hi ? ((t1 / 3) % 3) : ((t0 / 3) % 3);
        int dx = hi ? (t1 % 3) : (t0 % 3);
        bool tval = hi ? (t1 < 27) : true;    // t0 <= 23 always valid
        int iz = z0 + dz - 1, iy = y0 + dy - 1, ix = vx + dx - 1;
        _Float16 v = (_Float16)0;
        if (tval && (unsigned)iz < (unsigned)D && (unsigned)iy < (unsigned)D &&
            (unsigned)ix < (unsigned)D) {
            v = vbase[((ptrdiff_t)dz * D + dy) * D + dx];
        }
        A[j] = v;
    }
    v16h B = *(const v16h*)(wpk + (size_t)lane * 16);
    v8f acc = {0.f, 0.f, 0.f, 0.f, 0.f, 0.f, 0.f, 0.f};
    acc = __builtin_amdgcn_wmma_f32_16x16x32_f16(
        false, A, false, B, (short)0, acc, false, false);

    conv_epilogue<D, 16, 16, 0, 1, false>(acc, bias, mask, out, nullptr,
                                          z0, y0, tx, lane, 0);
}

// ---------------------------------------------------------------------------
// Host-side orchestration (one fully-constant instantiation per call site)
// ---------------------------------------------------------------------------
template <int KS, int DIN, int CIN, int COUT, int CSTRIDE, int CHOFS,
          int RELU, bool HASRES>
static void run_conv_t(hipStream_t st, const _Float16* in, const float* w,
                       const float* bias, const float* mask, _Float16* out,
                       const _Float16* res, _Float16*& wcur)
{
    constexpr int DOUT   = (KS == 2) ? DIN / 2 : DIN;
    constexpr int NT     = KS * KS * KS;
    constexpr int KTOT   = NT * CIN;
    constexpr int CHUNKS = (KTOT + 31) / 32;
    constexpr int N16S   = (COUT + 15) / 16;
    constexpr int TOTAL  = CHUNKS * N16S;

    pack_w_kernel<<<(TOTAL * 512 + 255) / 256, 256, 0, st>>>(
        w, wcur, KTOT, COUT, TOTAL, N16S);

    constexpr long THR = (long)DOUT * DOUT * (DOUT / 16) * N16S * 32;
    conv3d_wmma_kernel<KS, DIN, CIN, COUT, CSTRIDE, CHOFS, RELU, HASRES>
        <<<(unsigned)((THR + 255) / 256), 256, 0, st>>>(
            in, wcur, bias, mask, out, res);

    wcur += (size_t)TOTAL * 512;
}

// InceptionResNet block: y = concat(branch0(C/2), branch1(C/2)) + h
template <int D, int C>
static void run_irb_t(hipStream_t st, const _Float16* h, _Float16* y,
                      _Float16* t1, _Float16* t2, const float* m,
                      _Float16*& wcur, const float* const* pw)
{
    constexpr int C4 = C / 4, C2 = C / 2;
    run_conv_t<3, D, C,  C4, C4, 0,  1, false>(st, h,  pw[0], pw[1], m, t1, nullptr, wcur);
    run_conv_t<3, D, C4, C2, C,  0,  0, true >(st, t1, pw[2], pw[3], m, y,  h,       wcur);
    run_conv_t<1, D, C,  C4, C4, 0,  1, false>(st, h,  pw[4], pw[5], m, t1, nullptr, wcur);
    run_conv_t<3, D, C4, C4, C4, 0,  1, false>(st, t1, pw[6], pw[7], m, t2, nullptr, wcur);
    run_conv_t<1, D, C4, C2, C,  C2, 0, true >(st, t2, pw[8], pw[9], m, y,  h,       wcur);
}

extern "C" void kernel_launch(void* const* d_in, const int* in_sizes, int n_in,
                              void* d_out, int out_size, void* d_ws, size_t ws_size,
                              hipStream_t stream)
{
    (void)in_sizes; (void)out_size; (void)ws_size;
    const size_t D0 = 128, D1 = 64, D2 = 32, D3 = 16;
    const size_t V0 = D0*D0*D0, V1 = D1*D1*D1, V2 = D2*D2*D2, V3 = D3*D3*D3;

    // ---- inputs (setup_inputs dict insertion order: x, params..., mask) ----
    int c = 0;
    const float* x = (const float*)d_in[c++];
    const float* conv0_w = (const float*)d_in[c++];
    const float* conv0_b = (const float*)d_in[c++];
    const float* down0_w = (const float*)d_in[c++];
    const float* down0_b = (const float*)d_in[c++];
    const float* blk0[3][10];
    for (int b = 0; b < 3; ++b) for (int k = 0; k < 10; ++k) blk0[b][k] = (const float*)d_in[c++];
    const float* conv1_w = (const float*)d_in[c++];
    const float* conv1_b = (const float*)d_in[c++];
    const float* down1_w = (const float*)d_in[c++];
    const float* down1_b = (const float*)d_in[c++];
    const float* blk1[3][10];
    for (int b = 0; b < 3; ++b) for (int k = 0; k < 10; ++k) blk1[b][k] = (const float*)d_in[c++];
    const float* conv2_w = (const float*)d_in[c++];
    const float* conv2_b = (const float*)d_in[c++];
    const float* down2_w = (const float*)d_in[c++];
    const float* down2_b = (const float*)d_in[c++];
    const float* blk2[3][10];
    for (int b = 0; b < 3; ++b) for (int k = 0; k < 10; ++k) blk2[b][k] = (const float*)d_in[c++];
    const float* conv3_w = (const float*)d_in[c++];
    const float* conv3_b = (const float*)d_in[c++];
    const float* mask0   = (const float*)d_in[n_in - 1];

    // ---- workspace bump allocator ----
    char* wsp = (char*)d_ws;
    auto alloc = [&](size_t bytes) -> void* {
        void* p = (void*)wsp;
        wsp += (bytes + 255) & ~(size_t)255;
        return p;
    };
    _Float16* x16 = (_Float16*)alloc(V0 * 2);            // x as f16
    _Float16* big = (_Float16*)alloc(V0 * 16 * 2);       // conv0/conv1/conv2 outs
    _Float16* hA  = (_Float16*)alloc(V1 * 32 * 2);       // ping
    _Float16* hB  = (_Float16*)alloc(V1 * 32 * 2);       // pong
    _Float16* t1  = (_Float16*)alloc(V1 * 8 * 2);        // branch temp
    _Float16* t2  = (_Float16*)alloc(V1 * 8 * 2);        // branch temp
    float* m1 = (float*)alloc(V1 * 4);
    float* m2 = (float*)alloc(V2 * 4);
    float* m3 = (float*)alloc(V3 * 4);
    _Float16* wpk = (_Float16*)alloc(2u * 1024 * 1024 * 2); // packed weights
    _Float16* wcur = wpk;

    // ---- prep: x -> f16, mask pyramid ----
    f32_to_f16_kernel<<<(unsigned)((V0 + 255) / 256), 256, 0, stream>>>(x, x16, V0);
    pool_mask_kernel<<<(unsigned)((V1 + 255) / 256), 256, 0, stream>>>(mask0, m1, (int)D1);
    pool_mask_kernel<<<(unsigned)((V2 + 255) / 256), 256, 0, stream>>>(m1, m2, (int)D2);
    pool_mask_kernel<<<(unsigned)((V3 + 255) / 256), 256, 0, stream>>>(m2, m3, (int)D3);

    // ---- stage 0 ----
    {   // conv0: 3x3x3 SAME, 1->16 @ 128^3
        pack_w_kernel<<<2, 256, 0, stream>>>(conv0_w, wcur, 27, 16, 1, 1);
        constexpr long THR = 128L * 128 * 8 * 32;
        conv0_wmma_kernel<128><<<(unsigned)((THR + 255) / 256), 256, 0, stream>>>(
            x16, wcur, conv0_b, mask0, big);
        wcur += 512;
    }
    run_conv_t<2, 128, 16, 32, 32, 0, 1, false>(
        stream, big, down0_w, down0_b, m1, hA, nullptr, wcur);
    for (int b = 0; b < 3; ++b) {
        _Float16* h = (b & 1) ? hB : hA;
        _Float16* y = (b & 1) ? hA : hB;
        run_irb_t<64, 32>(stream, h, y, t1, t2, m1, wcur, blk0[b]);
    }
    f16_to_f32_kernel<<<(unsigned)((V1 * 32 + 255) / 256), 256, 0, stream>>>(
        hB, (float*)d_out, V1 * 32);            // out0 [64^3,32]

    // ---- stage 1 ----
    run_conv_t<3, 64, 32, 32, 32, 0, 1, false>(
        stream, hB, conv1_w, conv1_b, m1, big, nullptr, wcur);
    run_conv_t<2, 64, 32, 64, 64, 0, 1, false>(
        stream, big, down1_w, down1_b, m2, hA, nullptr, wcur);
    for (int b = 0; b < 3; ++b) {
        _Float16* h = (b & 1) ? hB : hA;
        _Float16* y = (b & 1) ? hA : hB;
        run_irb_t<32, 64>(stream, h, y, t1, t2, m2, wcur, blk1[b]);
    }
    f16_to_f32_kernel<<<(unsigned)((V2 * 64 + 255) / 256), 256, 0, stream>>>(
        hB, (float*)d_out + V1 * 32, V2 * 64);  // out1 [32^3,64]

    // ---- stage 2 ----
    run_conv_t<3, 32, 64, 64, 64, 0, 1, false>(
        stream, hB, conv2_w, conv2_b, m2, big, nullptr, wcur);
    run_conv_t<2, 32, 64, 32, 32, 0, 1, false>(
        stream, big, down2_w, down2_b, m3, hA, nullptr, wcur);
    for (int b = 0; b < 3; ++b) {
        _Float16* h = (b & 1) ? hB : hA;
        _Float16* y = (b & 1) ? hA : hB;
        run_irb_t<16, 32>(stream, h, y, t1, t2, m3, wcur, blk2[b]);
    }
    run_conv_t<3, 16, 32, 8, 8, 0, 0, false>(
        stream, hB, conv3_w, conv3_b, m3, t1, nullptr, wcur);
    f16_to_f32_kernel<<<(unsigned)((V3 * 8 + 255) / 256), 256, 0, stream>>>(
        t1, (float*)d_out + V1 * 32 + V2 * 64, V3 * 8); // out2 [16^3,8]
}
